// CAT_8349416423796
// MI455X (gfx1250) — matrix operand order repaired and checked by reading, not compiled, not run on hardware
//
#include <hip/hip_runtime.h>
#include <math.h>

typedef __attribute__((ext_vector_type(16))) _Float16 v16h;
typedef __attribute__((ext_vector_type(8)))  _Float16 v8h;
typedef __attribute__((ext_vector_type(8)))  float    v8f;

#define HW    512
#define NPIX  (HW * HW)        // 262144
#define NB    8
#define KTOP  26214            // int(0.1 * 262144)
#define NBIN  4096

// ---------------------------------------------------------------- utilities
__device__ __forceinline__ float sigf(float z) { return 1.f / (1.f + __expf(-z)); }
__device__ __forceinline__ int   refl(int i)   { return i < 0 ? -i : (i > HW - 1 ? 2 * (HW - 1) - i : i); }
__device__ __forceinline__ float safemax(float m) { return m > 0.f ? m : 1.f; }
__device__ __forceinline__ int bin1of(float v, float mx) {
    int b = (int)(v * (4096.f / safemax(mx)));
    return b < 0 ? 0 : (b > NBIN - 1 ? NBIN - 1 : b);
}
__device__ __forceinline__ int bin2of(float v, float mx, int b1) {
    float w = safemax(mx) * (1.f / 4096.f);
    int s = (int)((v - (float)b1 * w) * (4096.f / w));
    return s < 0 ? 0 : (s > NBIN - 1 ? NBIN - 1 : s);
}

// ---------------------------------------------------------------- mean over H,W per (b,c)
__global__ __launch_bounds__(256) void k_mean(const float* __restrict__ x, float* __restrict__ mean) {
    __shared__ float red[256];
    const int bc = blockIdx.x;                 // 0..23
    const float* p = x + (size_t)bc * NPIX;
    float s = 0.f;
    for (int i = threadIdx.x; i < NPIX; i += 256) s += p[i];
    red[threadIdx.x] = s;
    __syncthreads();
    for (int off = 128; off > 0; off >>= 1) {
        if (threadIdx.x < off) red[threadIdx.x] += red[threadIdx.x + off];
        __syncthreads();
    }
    if (threadIdx.x == 0) mean[bc] = red[0] * (1.f / (float)NPIX);
}

// ---------------------------------------------------------------- color = sigmoid(mean @ Wc^T + bc)
__global__ void k_color(const float* __restrict__ mean, const float* __restrict__ cw,
                        const float* __restrict__ cb, float* __restrict__ color) {
    int i = threadIdx.x;
    if (i < NB * 3) {
        int b = i / 3, j = i % 3;
        float z = cb[j];
        for (int c = 0; c < 3; ++c) z += mean[b * 3 + c] * cw[j * 3 + c];
        color[i] = sigf(z);
    }
}

// ---------------------------------------------------------------- fused conv1(WMMA)+conv2(WMMA)+epilogue
// One block = one 16x16 output tile of one batch image.
#define XP 20            // x tile with 2-halo
#define HP 18            // h tile with 1-halo (18x18 = 324 pixels)
#define HPP 336          // hs padded to 21 whole N-tiles
#define NT1 21           // 336 / 16
#define ZPAD (3 * XP * XP)   // zero slot in xs
__global__ __launch_bounds__(256) void k_conv_fused(
    const float* __restrict__ x,   const float* __restrict__ w1, const float* __restrict__ pb1,
    const float* __restrict__ w2,  const float* __restrict__ pb2,
    const float* __restrict__ color, const float* __restrict__ cw, const float* __restrict__ cb,
    float* __restrict__ xout)
{
    // hs is PIXEL-major: hs[p*64 + c]; conv2 K-order is k = (dy*3+dx)*64 + c
    __shared__ __attribute__((aligned(32))) _Float16 xs[3 * XP * XP + 16]; // x tile f16 + zero pad
    __shared__ __attribute__((aligned(32))) _Float16 hs[HPP * 64];         // 43008 B (padded)
    __shared__ __attribute__((aligned(32))) _Float16 w1s[64 * 32];         // w1, K padded 27->32
    __shared__ __attribute__((aligned(32))) _Float16 w2s[4 * 576];         // w2 permuted, row3 = 0
    __shared__ __attribute__((aligned(32))) float    b1s[64];
    __shared__ float b2s[3], cols[3], cws[12], cbs[4];

    const int tid  = threadIdx.x;
    const int lane = tid & 31;
    const int wave = tid >> 5;
    const int half = lane >> 4;
    const int lr   = lane & 15;

    const int bidx = blockIdx.z;
    const int gy0  = blockIdx.y * 16;
    const int gx0  = blockIdx.x * 16;

    // ---- stage weights / scalars into LDS
    for (int i = tid; i < 64 * 32; i += 256) {
        int m = i >> 5, k = i & 31;
        w1s[i] = (k < 27) ? (_Float16)w1[m * 27 + k] : (_Float16)0.f;
    }
    for (int i = tid; i < 4 * 576; i += 256) {     // permuted: k = dd*64 + c
        int m = i / 576, k = i % 576;
        int dd = k >> 6, c = k & 63;
        w2s[i] = (m < 3) ? (_Float16)w2[(m * 64 + c) * 9 + dd] : (_Float16)0.f;
    }
    if (tid < 64) b1s[tid] = pb1[tid];
    if (tid < 3)  { b2s[tid] = pb2[tid]; cols[tid] = color[bidx * 3 + tid]; }
    if (tid < 12) cws[tid] = cw[tid];
    if (tid < 4)  cbs[tid] = cb[tid];

    // ---- stage x tile (3 x 20 x 20), zero padded outside image; tail slots zeroed
    for (int i = tid; i < 3 * XP * XP + 16; i += 256) {
        float v = 0.f;
        if (i < 3 * XP * XP) {
            int c = i / (XP * XP), rem = i % (XP * XP);
            int ly = rem / XP, lx = rem % XP;
            int gy = gy0 - 2 + ly, gx = gx0 - 2 + lx;
            if (gy >= 0 && gy < HW && gx >= 0 && gx < HW) {
                const float* gp = &x[((size_t)(bidx * 3 + c) * HW + gy) * HW + gx];
                v = *gp;
                __builtin_prefetch(gp + HW, 0, 0);   // global_prefetch_b8
            }
        }
        xs[i] = (_Float16)v;
    }
    __syncthreads();

    // ============== conv1: h(64 x 324) = A(64x32) * B(32x324), relu -> LDS (pixel-major f16)
    // A fragment: lane row M=lr; elems 0..7 -> K = half*8+e, elems 8..15 -> K = 16+half*8+e
    v16h a1[4];
    v8f  bias1[4];
#pragma unroll
    for (int mt = 0; mt < 4; ++mt) {
        const _Float16* row = w1s + (mt * 16 + lr) * 32;
        v8h lo = *(const v8h*)(row + half * 8);
        v8h hi = *(const v8h*)(row + 16 + half * 8);
#pragma unroll
        for (int e = 0; e < 8; ++e) { a1[mt][e] = lo[e]; a1[mt][e + 8] = hi[e]; }
        bias1[mt] = *(const v8f*)(b1s + mt * 16 + half * 8);
    }
    for (int t = wave; t < NT1; t += 8) {
        // B fragment: lane column N=lr; elems 0..7 -> K = half*8+e, elems 8..15 -> K = 16+half*8+e (K-order: c*9+dy*3+dx)
        int p  = t * 16 + lr;                 // 0..335 (324..335 = padding pixels)
        int pc = (p < HP * HP) ? p : 0;       // clamp reads to valid range
        int py = pc / HP, px = pc % HP;
        v16h bfrag;
#pragma unroll
        for (int e = 0; e < 16; ++e) {
            int k = (e < 8) ? (half * 8 + e) : (16 + half * 8 + (e - 8));
            int c = k / 9, r9 = k % 9, dy = r9 / 3, dx = r9 % 3;
            int idx = c * (XP * XP) + (py + dy) * XP + (px + dx);
            bfrag[e] = xs[(k < 27) ? idx : ZPAD];    // index-select, unconditional load
        }
#pragma unroll
        for (int mt = 0; mt < 4; ++mt) {
            v8f acc = {};
            acc = __builtin_amdgcn_wmma_f32_16x16x32_f16(false, a1[mt], false, bfrag,
                                                         (short)0, acc, false, false);
            v8h hv;
#pragma unroll
            for (int r = 0; r < 8; ++r) {
                float v = acc[r] + bias1[mt][r];
                hv[r] = (_Float16)(v > 0.f ? v : 0.f);
            }
            *(v8h*)(hs + p * 64 + mt * 16 + half * 8) = hv;  // unconditional b128 store
        }
    }
    __syncthreads();

    // ============== conv2: D(16x256) = A2(16x576) * B2(576x256); K = dd*64 + c, 18 steps of 32
    // Wave handles N-tiles 2w and 2w+1: p = t*16+lr  ->  oy = t, ox = lr.
    const int oy0 = wave * 2;
    const int mi  = (lr < 3) ? lr : 3;        // row 3 of w2s is zeros
    const _Float16* arow = w2s + mi * 576;
    v8f acc0 = {}, acc1 = {};
#pragma unroll
    for (int s = 0; s < 18; ++s) {
        const int dd = s >> 1, dy = dd / 3, dx = dd % 3;   // compile-time per step
        const int cbase = (s & 1) * 32;
        // A fragment (shared by both N-tiles): two 16B LDS loads
        v8h alo = *(const v8h*)(arow + s * 32 + half * 16);
        v8h ahi = *(const v8h*)(arow + s * 32 + half * 16 + 8);
        v16h afrag;
#pragma unroll
        for (int e = 0; e < 8; ++e) { afrag[e] = alo[e]; afrag[e + 8] = ahi[e]; }
        // B fragments: 16 contiguous f16 (channels) -> one 32B LDS load each
        const _Float16* hb = hs + (lr + dx) * 64 + cbase + half * 16;
        v16h b0 = *(const v16h*)(hb + (oy0 + dy) * (HP * 64));
        v16h b1f = *(const v16h*)(hb + (oy0 + 1 + dy) * (HP * 64));
        acc0 = __builtin_amdgcn_wmma_f32_16x16x32_f16(false, afrag, false, b0,
                                                      (short)0, acc0, false, false);
        acc1 = __builtin_amdgcn_wmma_f32_16x16x32_f16(false, afrag, false, b1f,
                                                      (short)0, acc1, false, false);
    }
    // epilogue: out-channels 0..2 live in lanes 0..15, elements 0..2
    if (half == 0) {
#pragma unroll
        for (int q = 0; q < 2; ++q) {
            v8f acc = q ? acc1 : acc0;
            int oy = oy0 + q;
            float i0 = sigf(acc[0] + b2s[0]) * cols[0];
            float i1 = sigf(acc[1] + b2s[1]) * cols[1];
            float i2 = sigf(acc[2] + b2s[2]) * cols[2];
            float Kf = cws[0] * i0 + cws[1] * i1 + cws[2] * i2 + cbs[0];
            int gy = gy0 + oy, gx = gx0 + lr;
#pragma unroll
            for (int c = 0; c < 3; ++c) {
                float Bt = cws[(1 + c) * 3 + 0] * i0 + cws[(1 + c) * 3 + 1] * i1 +
                           cws[(1 + c) * 3 + 2] * i2 + cbs[1 + c];
                size_t gi = ((size_t)(bidx * 3 + c) * HW + gy) * HW + gx;
                float xv = x[gi];
                xout[gi] = (Kf + 1.f) * xv - Bt;
            }
        }
    }
}

// ---------------------------------------------------------------- zero scratch (hist/max/partials)
__global__ void k_zero(unsigned int* __restrict__ p, int n) {
    int i = blockIdx.x * blockDim.x + threadIdx.x;
    if (i < n) p[i] = 0u;
}

// ---------------------------------------------------------------- bright map + per-batch max
__global__ __launch_bounds__(256) void k_bright(const float* __restrict__ xout,
                                                float* __restrict__ bright,
                                                unsigned int* __restrict__ bmax) {
    int i = blockIdx.x * 256 + threadIdx.x;
    if (i >= NB * NPIX) return;
    int b = i >> 18, rem = i & (NPIX - 1), y = rem >> 9, xq = rem & (HW - 1);
    const float* base = xout + (size_t)b * 3 * NPIX;
    float s = 0.f;
#pragma unroll
    for (int dy = -1; dy <= 1; ++dy) {
#pragma unroll
        for (int dx = -1; dx <= 1; ++dx) {
            int o = refl(y + dy) * HW + refl(xq + dx);
            float m = base[o];
            float m1 = base[NPIX + o];     if (m1 > m) m = m1;
            float m2 = base[2 * NPIX + o]; if (m2 > m) m = m2;
            s += (dy == 0 && dx == 0) ? m : m * (1.f / 9.f);
        }
    }
    s = s > 0.f ? s : 0.f;
    bright[i] = s;
    atomicMax(&bmax[b], __float_as_uint(s));   // s >= 0 -> uint order == float order
}

// ---------------------------------------------------------------- level-1 / level-2 histograms
__global__ __launch_bounds__(256) void k_hist1(const float* __restrict__ bright,
                                               const unsigned int* __restrict__ bmax,
                                               int* __restrict__ hist) {
    int i = blockIdx.x * 256 + threadIdx.x;
    if (i >= NB * NPIX) return;
    int b = i >> 18;
    float mx = __uint_as_float(bmax[b]);
    atomicAdd(&hist[b * NBIN + bin1of(bright[i], mx)], 1);
}

__global__ __launch_bounds__(256) void k_hist2(const float* __restrict__ bright,
                                               const unsigned int* __restrict__ bmax,
                                               const int* __restrict__ b1a,
                                               int* __restrict__ hist) {
    int i = blockIdx.x * 256 + threadIdx.x;
    if (i >= NB * NPIX) return;
    int b = i >> 18;
    float mx = __uint_as_float(bmax[b]);
    float v = bright[i];
    int b1 = b1a[b];
    if (bin1of(v, mx) == b1)
        atomicAdd(&hist[b * NBIN + bin2of(v, mx, b1)], 1);
}

// find largest bin with suffix-count >= K; report count strictly above it
__global__ void k_scan(const int* __restrict__ hist, const int* __restrict__ kin, int kconst,
                       int* __restrict__ b_out, int* __restrict__ need_out) {
    int b = blockIdx.x;
    if (threadIdx.x != 0) return;
    int K = kin ? kin[b] : kconst;
    const int* h = hist + b * NBIN;
    long cum = 0, above = 0;
    int bsel = 0;
    for (int i = NBIN - 1; i >= 0; --i) {
        cum += h[i];
        if (cum >= K) { bsel = i; above = cum - h[i]; break; }
    }
    b_out[b] = bsel;
    if (need_out) need_out[b] = K - (int)above;
}

// ---------------------------------------------------------------- deterministic pick + partial sums
__global__ __launch_bounds__(256) void k_pick(const float* __restrict__ xout,
                                              const float* __restrict__ bright,
                                              const unsigned int* __restrict__ bmax,
                                              const int* __restrict__ b1a,
                                              const int* __restrict__ b2a,
                                              float* __restrict__ partials) {
    __shared__ float r0[256], r1[256], r2[256], rc[256];
    const int b = blockIdx.y, blk = blockIdx.x, t = threadIdx.x;
    const float mx = __uint_as_float(bmax[b]);
    const int b1 = b1a[b], b2 = b2a[b];
    const float* bflat = bright + (size_t)b * NPIX;
    const float* xflat = xout + (size_t)b * 3 * NPIX;
    float s0 = 0.f, s1 = 0.f, s2 = 0.f, cn = 0.f;
    for (int i = blk * 256 + t; i < NPIX; i += 512 * 256) {
        float v = bflat[i];
        int bin = bin1of(v, mx);
        bool acc = bin > b1;
        if (bin == b1) acc = bin2of(v, mx, b1) >= b2;
        if (acc) {
            // reference reshapes (B,C,H,W) -> (B,H*W,C): picked triple = flat[3i..3i+2]
            size_t off = (size_t)3 * i;
            s0 += xflat[off]; s1 += xflat[off + 1]; s2 += xflat[off + 2];
            cn += 1.f;
        }
    }
    r0[t] = s0; r1[t] = s1; r2[t] = s2; rc[t] = cn;
    __syncthreads();
    for (int off = 128; off > 0; off >>= 1) {
        if (t < off) { r0[t] += r0[t + off]; r1[t] += r1[t + off]; r2[t] += r2[t + off]; rc[t] += rc[t + off]; }
        __syncthreads();
    }
    if (t == 0) {
        float* p = partials + ((size_t)b * 512 + blk) * 4;
        p[0] = r0[0]; p[1] = r1[0]; p[2] = r2[0]; p[3] = rc[0];
    }
}

__global__ __launch_bounds__(256) void k_finalA(const float* __restrict__ partials,
                                                float* __restrict__ A) {
    __shared__ float r0[256], r1[256], r2[256], rc[256];
    const int b = blockIdx.x, t = threadIdx.x;
    float s0 = 0.f, s1 = 0.f, s2 = 0.f, cn = 0.f;
    for (int i = t; i < 512; i += 256) {
        const float* p = partials + ((size_t)b * 512 + i) * 4;
        s0 += p[0]; s1 += p[1]; s2 += p[2]; cn += p[3];
    }
    r0[t] = s0; r1[t] = s1; r2[t] = s2; rc[t] = cn;
    __syncthreads();
    for (int off = 128; off > 0; off >>= 1) {
        if (t < off) { r0[t] += r0[t + off]; r1[t] += r1[t + off]; r2[t] += r2[t + off]; rc[t] += rc[t + off]; }
        __syncthreads();
    }
    if (t == 0) {
        float inv = 1.f / (rc[0] > 0.f ? rc[0] : 1.f);
        A[b * 3 + 0] = r0[0] * inv;
        A[b * 3 + 1] = r1[0] * inv;
        A[b * 3 + 2] = r2[0] * inv;
    }
}

// ---------------------------------------------------------------- launch
extern "C" void kernel_launch(void* const* d_in, const int* in_sizes, int n_in,
                              void* d_out, int out_size, void* d_ws, size_t ws_size,
                              hipStream_t stream) {
    (void)in_sizes; (void)n_in; (void)out_size; (void)ws_size;
    const float* x  = (const float*)d_in[0];
    const float* w1 = (const float*)d_in[1];
    const float* b1 = (const float*)d_in[2];
    const float* w2 = (const float*)d_in[3];
    const float* b2 = (const float*)d_in[4];
    const float* cw = (const float*)d_in[5];   // color_w (3,3)
    const float* cb = (const float*)d_in[6];   // color_b (3)
    const float* vw = (const float*)d_in[7];   // conv_w  (4,3,1,1)
    const float* vb = (const float*)d_in[8];   // conv_b  (4)

    float* xout = (float*)d_out;                       // 8*3*512*512
    float* Aout = xout + (size_t)NB * 3 * NPIX;        // 24

    float* ws      = (float*)d_ws;
    float* mean    = ws;                               // 24
    float* color   = ws + 24;                          // 24
    float* bright  = ws + 64;                          // 8*262144
    unsigned int* bmax = (unsigned int*)(bright + (size_t)NB * NPIX);  // 8
    int*   b1i     = (int*)(bmax + NB);                // 8
    int*   needi   = b1i + NB;                         // 8
    int*   b2i     = needi + NB;                       // 8
    int*   hist1   = b2i + NB;                         // 8*4096
    int*   hist2   = hist1 + NB * NBIN;                // 8*4096
    float* parts   = (float*)(hist2 + NB * NBIN);      // 8*512*4

    k_mean<<<NB * 3, 256, 0, stream>>>(x, mean);
    k_color<<<1, 32, 0, stream>>>(mean, cw, cb, color);

    dim3 gconv(HW / 16, HW / 16, NB);
    k_conv_fused<<<gconv, 256, 0, stream>>>(x, w1, b1, w2, b2, color, vw, vb, xout);

    int zn = NB + 3 * NB + 2 * NB * NBIN + NB * 512 * 4;   // bmax..partials contiguous
    k_zero<<<(zn + 255) / 256, 256, 0, stream>>>(bmax, zn);

    int nblk = (NB * NPIX + 255) / 256;
    k_bright<<<nblk, 256, 0, stream>>>(xout, bright, bmax);
    k_hist1<<<nblk, 256, 0, stream>>>(bright, bmax, hist1);
    k_scan<<<NB, 32, 0, stream>>>(hist1, nullptr, KTOP, b1i, needi);
    k_hist2<<<nblk, 256, 0, stream>>>(bright, bmax, b1i, hist2);
    k_scan<<<NB, 32, 0, stream>>>(hist2, needi, 0, b2i, nullptr);
    k_pick<<<dim3(512, NB), 256, 0, stream>>>(xout, bright, bmax, b1i, b2i, parts);
    k_finalA<<<NB, 256, 0, stream>>>(parts, Aout);
}
